// CSHead2_32882269618327
// MI455X (gfx1250) — compile-verified, matrix operand order checked by reading, not run on hardware
//
#include <hip/hip_runtime.h>

// Problem constants (match reference)
#define NB    8
#define HH    112
#define WW    112
#define CCH   512
#define NOBJ  64
#define NPART 12
#define NE    16
#define WT    (WW / 16)             // 7 w-tiles per row
#define HT    (HH / 4)              // 28 h-supertiles (4 rows each)
#define SPTC  (WT * HT * NB)        // 1568 conv blocks per COUT-slab
#define SPTH  (WT * HH * NB)        // 6272 head spatial tiles

typedef __attribute__((ext_vector_type(16))) __bf16 v16bf;
typedef __attribute__((ext_vector_type(8)))  float  v8f;

union BF16x16 { unsigned int u[8]; v16bf v; };

__device__ __forceinline__ unsigned short f2bf(float f) {
    unsigned int u = __float_as_uint(f);
    u += 0x7FFFu + ((u >> 16) & 1u);          // round to nearest even
    return (unsigned short)(u >> 16);
}
__device__ __forceinline__ float bf2f(unsigned short h) {
    return __uint_as_float(((unsigned int)h) << 16);
}
__device__ __forceinline__ unsigned int pack2(float a, float b) {
    return (unsigned int)f2bf(a) | ((unsigned int)f2bf(b) << 16);
}
// WMMA 16x16x32 bf16 per-lane K offsets (ISA 7.12.2 16-bit A layout)
__device__ __forceinline__ int kofs_of(int lane, int j) {
    return ((j >= 4) ? 16 : 0) + ((lane >> 4) << 3) + ((j & 3) << 1);
}

// CDNA5 async global->LDS copy (ASYNCcnt-tracked). l = LDS byte offset.
__device__ __forceinline__ void async_ld_b32(unsigned long long g, unsigned int l) {
    asm volatile("global_load_async_to_lds_b32 %0, %1, off"
                 :: "v"(l), "v"(g) : "memory");
}
__device__ __forceinline__ void async_wait0() {
    asm volatile("s_wait_asynccnt 0x0" ::: "memory");
}

// ---------------------------------------------------------------------------
// Weight packers -> bf16 WMMA B-fragment layout:
//   u32 index = ((chunk * ntiles + nt) * 32 + lane) * 8 + j
//   value     = pack2( W[n][k], W[n][k+1] ),  n = nt*16 + (lane&15),
//               k = chunk_cin_base + kofs_of(lane, j)
// ---------------------------------------------------------------------------
__global__ void pack_conv_w(const float* __restrict__ w,    // [cout][cin][3][3]
                            unsigned int* __restrict__ out,
                            int cin, int cout) {
    const int ccn = cin >> 5, ntl = cout >> 4;
    long idx = (long)blockIdx.x * blockDim.x + threadIdx.x;
    long total = (long)9 * ccn * ntl * 256;
    if (idx >= total) return;
    int j    = idx & 7;
    int lane = (idx >> 3) & 31;
    long r   = idx >> 8;
    int nt = (int)(r % ntl); r /= ntl;
    int cc = (int)(r % ccn); r /= ccn;
    int ky = (int)r / 3, kx = (int)r % 3;
    int n = nt * 16 + (lane & 15);
    int k = cc * 32 + kofs_of(lane, j);
    long b0 = (((long)n * cin + k) * 3 + ky) * 3 + kx;
    long b1 = (((long)n * cin + k + 1) * 3 + ky) * 3 + kx;
    out[idx] = pack2(w[b0], w[b1]);
}

__global__ void pack_head_w(const float* __restrict__ w,    // [nout][cin]
                            unsigned int* __restrict__ out,
                            int cin, int nout, int ntl) {
    long idx = (long)blockIdx.x * blockDim.x + threadIdx.x;
    long total = (long)(cin >> 5) * ntl * 256;
    if (idx >= total) return;
    int j = idx & 7, lane = (idx >> 3) & 31;
    long r = idx >> 8;
    int nt = (int)(r % ntl);
    int cc = (int)(r / ntl);
    int n = nt * 16 + (lane & 15);
    int k = cc * 32 + kofs_of(lane, j);
    if (n >= nout) { out[idx] = 0u; return; }
    out[idx] = pack2(w[(long)n * cin + k], w[(long)n * cin + k + 1]);
}

__global__ void pack_part_w(const float* __restrict__ wheads,  // [E][P][C]
                            const int* __restrict__ inst,
                            unsigned int* __restrict__ out) {  // [B][16][256]
    int idx = blockIdx.x * blockDim.x + threadIdx.x;
    if (idx >= NB * 16 * 256) return;
    int j = idx & 7, lane = (idx >> 3) & 31;
    int cc = (idx >> 8) & 15;
    int b  = idx >> 12;
    int n = lane & 15;
    int k = cc * 32 + kofs_of(lane, j);
    if (n >= NPART) { out[idx] = 0u; return; }
    const float* wb = wheads + ((long)inst[b] * NPART + n) * CCH;
    out[idx] = pack2(wb[k], wb[k + 1]);
}

__global__ void gather_part_b(const float* __restrict__ bheads, // [E][P]
                              const int* __restrict__ inst,
                              float* __restrict__ out) {        // [B][16]
    int i = blockIdx.x * blockDim.x + threadIdx.x;
    if (i >= NB * 16) return;
    int b = i >> 4, n = i & 15;
    out[i] = (n < NPART) ? bheads[inst[b] * NPART + n] : 0.f;
}

// ---------------------------------------------------------------------------
// x = features * emb[row(b)] (+ optional NHWC bf16 addend) -> NHWC bf16
// ---------------------------------------------------------------------------
__global__ void scale_embed(const float* __restrict__ feat,        // NCHW fp32
                            const float* __restrict__ emb,         // [E+1][C]
                            const int* __restrict__ inst,          // null -> row E
                            const unsigned short* __restrict__ add_nhwc,
                            unsigned short* __restrict__ out_nhwc) {
    size_t i = (size_t)blockIdx.x * blockDim.x + threadIdx.x;
    const size_t total = (size_t)NB * HH * WW * CCH;
    if (i >= total) return;
    int c = (int)(i & (CCH - 1));
    size_t s = i >> 9;
    int b = (int)(s / (HH * WW));
    int hw = (int)(s % (HH * WW));
    int row = inst ? inst[b] : NE;
    float v = feat[((size_t)b * CCH + c) * (HH * WW) + hw] * emb[(size_t)row * CCH + c];
    if (add_nhwc) v += bf2f(add_nhwc[i]);
    out_nhwc[i] = f2bf(v);
}

// ---------------------------------------------------------------------------
// 3x3 SAME conv, implicit GEMM, v_wmma_f32_16x16x32_bf16.
// Block = 256 thr = 8 waves; each wave: 16 out channels x (4 rows x 16 cols).
// Double-buffered LDS slab (6x18x32ch, pair-dim padded to 17): chunk cc+1 is
// fetched with async global->LDS copies while chunk cc runs 36 WMMA/wave.
// OOB halo: global address clamped, LDS dst -> trash word, real slot
// pre-zeroed once. One barrier per cin-chunk, no divergent staging branches.
// ---------------------------------------------------------------------------
#define SLAB  (6 * 18 * 17)
#define BUFU  (SLAB + 8)            // u32 stride between the two buffers

template <int CIN, int COUT, bool RELU>
__global__ __launch_bounds__(256, 1)
void conv3x3_wmma(const unsigned short* __restrict__ x,
                  const unsigned int* __restrict__ wfrag,
                  const float* __restrict__ bias,
                  unsigned short* __restrict__ y,
                  int y_cstride, int y_cbase) {
    __shared__ unsigned int lA[2 * BUFU];

    const int t  = blockIdx.x;
    const int w0 = (t % WT) * 16;
    const int h0 = ((t / WT) % HT) * 4;
    const int b  =  t / (WT * HT);

    const int tid  = threadIdx.x;
    const int lane = tid & 31;
    const int wave = tid >> 5;
    const int ntl  = COUT >> 4;
    const int ccn  = CIN >> 5;
    const int nt   = blockIdx.y * 8 + wave;

    const int mA = lane & 15;
    int kp[8];
#pragma unroll
    for (int j = 0; j < 8; ++j) kp[j] = kofs_of(lane, j) >> 1;

    const unsigned short* xin = x + ((long)b * HH * WW) * CIN;

    // ---- precompute staged-element descriptors (7 per thread, once) ----
    unsigned long long ge[7];   // clamped global address (advances +64B/chunk)
    unsigned int       le[7];   // LDS byte offset (trash for OOB/inactive)
#pragma unroll
    for (int it = 0; it < 7; ++it) {
        int i   = tid + it * 256;
        int pr  = i & 15;
        int col = (i >> 4) % 18;
        int row = (i >> 4) / 18;
        int hin = h0 + row - 1, win = w0 + col - 1;
        bool valid = (i < 6 * 18 * 16) &&
                     (hin >= 0) && (hin < HH) && (win >= 0) && (win < WW);
        int hc = min(max(hin, 0), HH - 1);
        int wc = min(max(win, 0), WW - 1);
        ge[it] = (unsigned long long)(size_t)(xin + ((long)hc * WW + wc) * CIN + pr * 2);
        unsigned real = (unsigned)((row * 18 + col) * 17 + pr);
        le[it] = (unsigned)(size_t)&lA[valid ? real : (unsigned)SLAB];
        if ((i < 6 * 18 * 16) && !valid) {      // pre-zero halo slots, both buffers
            lA[real] = 0u;
            lA[BUFU + real] = 0u;
        }
    }

    auto stage = [&](int par) {                 // issue async copies for next chunk
        const unsigned add = (unsigned)(par * (BUFU * 4));
#pragma unroll
        for (int it = 0; it < 7; ++it) {
            async_ld_b32(ge[it], le[it] + add);
            ge[it] += 64;                       // advance one cin-chunk
        }
    };

    v8f acc0 = {}, acc1 = {}, acc2 = {}, acc3 = {};
    const unsigned int* wq = wfrag + (size_t)nt * 256 + lane * 8;
    const size_t cstep = (size_t)ntl * 256;     // u32 per cin-chunk
    const size_t kstep = (size_t)ccn * cstep;   // u32 per (ky,kx)

    auto compute = [&](int cc, int par) {
        const int base = par * BUFU;
#pragma unroll
        for (int kk = 0; kk < 9; ++kk) {
            const int ky = kk / 3, kx = kk % 3;
            const uint4* wv = (const uint4*)(wq + (size_t)kk * kstep + (size_t)cc * cstep);
            uint4 q0 = wv[0], q1 = wv[1];
            BF16x16 bf;
            bf.u[0] = q0.x; bf.u[1] = q0.y; bf.u[2] = q0.z; bf.u[3] = q0.w;
            bf.u[4] = q1.x; bf.u[5] = q1.y; bf.u[6] = q1.z; bf.u[7] = q1.w;
#pragma unroll
            for (int r = 0; r < 4; ++r) {
                BF16x16 a;
#pragma unroll
                for (int j = 0; j < 8; ++j)
                    a.u[j] = lA[base + ((r + ky) * 18 + mA + kx) * 17 + kp[j]];
                v8f& A = (r == 0 ? acc0 : r == 1 ? acc1 : r == 2 ? acc2 : acc3);
                A = __builtin_amdgcn_wmma_f32_16x16x32_bf16(
                    false, a.v, false, bf.v, (short)0, A, false, false);
            }
        }
    };

    // ---- software pipeline: stage(cc+1) overlaps compute(cc) ----
    stage(0);
    async_wait0();
    __syncthreads();
    for (int cc = 0; cc < ccn; cc += 2) {       // ccn is even (2 or 16)
        if (cc + 1 < ccn) stage(1);
        compute(cc, 0);
        async_wait0();
        __syncthreads();
        if (cc + 2 < ccn) stage(0);
        compute(cc + 1, 1);
        async_wait0();
        __syncthreads();
    }

    // C/D layout: VGPR rr, lane -> M = rr + 8*(lane>>4), N = lane&15
    const int nc = nt * 16 + (lane & 15);
    const float bb = bias[nc];
    const int mb = (lane >> 4) * 8;
#pragma unroll
    for (int r = 0; r < 4; ++r) {
        v8f A = (r == 0 ? acc0 : r == 1 ? acc1 : r == 2 ? acc2 : acc3);
        unsigned short* yo = y
            + ((size_t)(b * HH + h0 + r) * WW + w0) * y_cstride + y_cbase + nc;
#pragma unroll
        for (int rr = 0; rr < 8; ++rr) {
            float v = A[rr] + bb;
            if (RELU) v = fmaxf(v, 0.f);
            yo[(size_t)(mb + rr) * y_cstride] = f2bf(v);
        }
    }
}

// ---------------------------------------------------------------------------
// 1x1 head GEMM (WMMA). x: NHWC bf16 [B,H,W,512]. Block: 128 thr = 4 waves.
// Writes fp32 NCHW (first nout_store channels) and optionally bf16 NHWC.
// ---------------------------------------------------------------------------
__global__ __launch_bounds__(128, 1)
void head_wmma(const unsigned short* __restrict__ x,
               const unsigned int* __restrict__ wfrag,
               const float* __restrict__ bias,
               float* __restrict__ out_nchw,
               unsigned short* __restrict__ out_nhwc,
               int nout_store, int nhwc_cs,
               long wfrag_bstride, int bias_bstride, int ntiles) {
    __shared__ unsigned int lA[16 * 17];

    const int t  = blockIdx.x;
    const int w0 = (t % WT) * 16;
    const int h  = (t / WT) % HH;
    const int b  =  t / (WT * HH);
    const int tid = threadIdx.x, lane = tid & 31, wave = tid >> 5;
    const int mA = lane & 15;
    int kp[8];
#pragma unroll
    for (int j = 0; j < 8; ++j) kp[j] = kofs_of(lane, j) >> 1;

    const unsigned short* xs = x + ((size_t)(b * HH + h) * WW + w0) * CCH;
    const unsigned int* wb = wfrag + (size_t)b * wfrag_bstride;

    const int m0 = tid >> 4,         p0 = tid & 15;   // elems 0..127
    const int m1 = (tid + 128) >> 4, p1 = tid & 15;   // elems 128..255

    v8f acc = {};
    for (int cc = 0; cc < (CCH >> 5); ++cc) {
        __syncthreads();
        lA[m0 * 17 + p0] = *(const unsigned int*)(xs + (size_t)m0 * CCH + cc * 32 + p0 * 2);
        lA[m1 * 17 + p1] = *(const unsigned int*)(xs + (size_t)m1 * CCH + cc * 32 + p1 * 2);
        __syncthreads();
        if (wave < ntiles) {
            BF16x16 a, bf;
#pragma unroll
            for (int j = 0; j < 8; ++j) a.u[j] = lA[mA * 17 + kp[j]];
            const uint4* wv =
                (const uint4*)(wb + ((size_t)cc * ntiles + wave) * 256 + lane * 8);
            uint4 q0 = wv[0], q1 = wv[1];
            bf.u[0] = q0.x; bf.u[1] = q0.y; bf.u[2] = q0.z; bf.u[3] = q0.w;
            bf.u[4] = q1.x; bf.u[5] = q1.y; bf.u[6] = q1.z; bf.u[7] = q1.w;
            acc = __builtin_amdgcn_wmma_f32_16x16x32_bf16(
                false, a.v, false, bf.v, (short)0, acc, false, false);
        }
    }

    if (wave < ntiles) {
        const int nc = wave * 16 + (lane & 15);
        const float bb = bias[bias_bstride * b + nc];
        const int mb = (lane >> 4) * 8;
#pragma unroll
        for (int r = 0; r < 8; ++r) {
            float v = acc[r] + bb;
            int wp = w0 + mb + r;
            if (nc < nout_store)
                out_nchw[(((size_t)b * nout_store + nc) * HH + h) * WW + wp] = v;
            if (out_nhwc)
                out_nhwc[((size_t)(b * HH + h) * WW + wp) * nhwc_cs + nc] = f2bf(v);
        }
    }
}

// ---------------------------------------------------------------------------
static inline void* wsal(char*& p, size_t bytes) {
    void* r = (void*)p;
    p += (bytes + 255) & ~(size_t)255;
    return r;
}

extern "C" void kernel_launch(void* const* d_in, const int* in_sizes, int n_in,
                              void* d_out, int out_size, void* d_ws, size_t ws_size,
                              hipStream_t stream) {
    const float* features = (const float*)d_in[0];
    const int*   inst     = (const int*)  d_in[1];
    const float* emb      = (const float*)d_in[2];
    const float* w_td1    = (const float*)d_in[3];
    const float* b_td1    = (const float*)d_in[4];
    const float* w_td2    = (const float*)d_in[5];
    const float* b_td2    = (const float*)d_in[6];
    const float* w_hobj   = (const float*)d_in[7];
    const float* b_hobj   = (const float*)d_in[8];
    const float* w_heads  = (const float*)d_in[9];
    const float* b_heads  = (const float*)d_in[10];
    const float* w_bus    = (const float*)d_in[11];
    const float* b_bus    = (const float*)d_in[12];
    const float* w_lat1   = (const float*)d_in[13];
    const float* b_lat1   = (const float*)d_in[14];
    const float* w_lat2   = (const float*)d_in[15];
    const float* b_lat2   = (const float*)d_in[16];
    const float* w_bu1    = (const float*)d_in[17];
    const float* b_bu1    = (const float*)d_in[18];
    const float* w_bu2    = (const float*)d_in[19];
    const float* b_bu2    = (const float*)d_in[20];

    float* out_obj  = (float*)d_out;                              // [B,64,H,W]
    float* out_part = out_obj + (size_t)NB * NOBJ * HH * WW;      // [B,12,H,W]

    const size_t ABUF = (size_t)NB * HH * WW * CCH * 2;
    const size_t OBUF = (size_t)NB * HH * WW * NOBJ * 2;
    char* p = (char*)d_ws;
    unsigned short* bx    = (unsigned short*)wsal(p, ABUF);
    unsigned short* bt1   = (unsigned short*)wsal(p, ABUF);
    unsigned short* bt2   = (unsigned short*)wsal(p, ABUF);
    unsigned short* bcat1 = (unsigned short*)wsal(p, ABUF);
    unsigned short* bcat2 = (unsigned short*)wsal(p, ABUF);
    unsigned short* bbu   = (unsigned short*)wsal(p, ABUF);
    unsigned short* bobj  = (unsigned short*)wsal(p, OBUF);
    unsigned int* wf_td1  = (unsigned int*)wsal(p, (size_t)18 * 512 * 512);
    unsigned int* wf_td2  = (unsigned int*)wsal(p, (size_t)18 * 512 * 512);
    unsigned int* wf_bus  = (unsigned int*)wsal(p, (size_t)18 * 64 * 256);
    unsigned int* wf_lat1 = (unsigned int*)wsal(p, (size_t)18 * 512 * 256);
    unsigned int* wf_lat2 = (unsigned int*)wsal(p, (size_t)18 * 512 * 256);
    unsigned int* wf_bu1  = (unsigned int*)wsal(p, (size_t)18 * 512 * 256);
    unsigned int* wf_bu2  = (unsigned int*)wsal(p, (size_t)18 * 512 * 512);
    unsigned int* wf_hobj = (unsigned int*)wsal(p, (size_t)16 * 4 * 256 * 4);
    unsigned int* wf_part = (unsigned int*)wsal(p, (size_t)NB * 16 * 256 * 4);
    float* bias_part      = (float*)wsal(p, NB * 16 * sizeof(float));

    auto packc = [&](const float* w, unsigned int* o, int cin, int cout) {
        long total = (long)9 * (cin >> 5) * (cout >> 4) * 256;
        pack_conv_w<<<(unsigned)((total + 255) / 256), 256, 0, stream>>>(w, o, cin, cout);
    };
    packc(w_td1, wf_td1, 512, 512);
    packc(w_td2, wf_td2, 512, 512);
    packc(w_bus, wf_bus, 64, 256);
    packc(w_lat1, wf_lat1, 512, 256);
    packc(w_lat2, wf_lat2, 512, 256);
    packc(w_bu1, wf_bu1, 512, 256);
    packc(w_bu2, wf_bu2, 512, 512);
    pack_head_w<<<64, 256, 0, stream>>>(w_hobj, wf_hobj, 512, NOBJ, 4);
    pack_part_w<<<(NB * 16 * 256 + 255) / 256, 256, 0, stream>>>(w_heads, inst, wf_part);
    gather_part_b<<<1, 128, 0, stream>>>(b_heads, inst, bias_part);

    const size_t elems = (size_t)NB * HH * WW * CCH;
    const unsigned sblocks = (unsigned)((elems + 255) / 256);

    // ---- top-down object pass ----
    scale_embed<<<sblocks, 256, 0, stream>>>(features, emb, nullptr, nullptr, bx);
    conv3x3_wmma<512, 512, true><<<dim3(SPTC, 4), 256, 0, stream>>>(bx, wf_td1, b_td1, bt1, 512, 0);
    conv3x3_wmma<512, 512, true><<<dim3(SPTC, 4), 256, 0, stream>>>(bt1, wf_td2, b_td2, bt2, 512, 0);
    head_wmma<<<SPTH, 128, 0, stream>>>(bt2, wf_hobj, b_hobj, out_obj, bobj,
                                        NOBJ, NOBJ, 0L, 0, 4);

    // ---- bottom-up pass (concat is free via channel-half writes) ----
    conv3x3_wmma<64, 256, false><<<dim3(SPTC, 2), 256, 0, stream>>>(bobj, wf_bus, b_bus, bcat1, 512, 0);
    conv3x3_wmma<512, 256, true><<<dim3(SPTC, 2), 256, 0, stream>>>(bt2, wf_lat1, b_lat1, bcat1, 512, 256);
    conv3x3_wmma<512, 256, true><<<dim3(SPTC, 2), 256, 0, stream>>>(bcat1, wf_bu1, b_bu1, bcat2, 512, 0);
    conv3x3_wmma<512, 256, true><<<dim3(SPTC, 2), 256, 0, stream>>>(bt1, wf_lat2, b_lat2, bcat2, 512, 256);
    conv3x3_wmma<512, 512, true><<<dim3(SPTC, 4), 256, 0, stream>>>(bcat2, wf_bu2, b_bu2, bbu, 512, 0);

    // ---- routed part pass (reuse bx/bt1/bt2) ----
    scale_embed<<<sblocks, 256, 0, stream>>>(features, emb, inst, bbu, bx);
    conv3x3_wmma<512, 512, true><<<dim3(SPTC, 4), 256, 0, stream>>>(bx, wf_td1, b_td1, bt1, 512, 0);
    conv3x3_wmma<512, 512, true><<<dim3(SPTC, 4), 256, 0, stream>>>(bt1, wf_td2, b_td2, bt2, 512, 0);
    head_wmma<<<SPTH, 128, 0, stream>>>(bt2, wf_part, bias_part, out_part, nullptr,
                                        NPART, 0, (long)16 * 256, 16, 1);
}
